// BitResNet50CIFAR_30116310680088
// MI455X (gfx1250) — matrix operand order repaired and checked by reading, not compiled, not run on hardware
//
#include <hip/hip_runtime.h>
#include <cstdint>
#include <cstddef>

// ---------------------------------------------------------------------------
// Types for CDNA5 WMMA (wave32): A,B = 16 halves/lane (8 VGPRs), C/D = 8 floats
// ---------------------------------------------------------------------------
typedef __attribute__((ext_vector_type(16))) _Float16 v16h;
typedef __attribute__((ext_vector_type(8)))  float    v8f;
typedef __attribute__((ext_vector_type(4)))  unsigned v4u;
// TDM descriptor group types (probe-verified builtin signature)
typedef __attribute__((ext_vector_type(4))) unsigned tdm4u;
typedef __attribute__((ext_vector_type(8))) int      tdm8i;
typedef __attribute__((ext_vector_type(4))) int      tdm4i;

#define BN_EPS 1e-5f
#define HBINS  2048

union AFrag { v16h v; _Float16 h[16]; v4u q[2]; };

static inline __host__ __device__ long hmin_l(long a, long b) { return a < b ? a : b; }

// ---------------------------------------------------------------------------
// Generic implicit-GEMM convolution (any KxK / stride / pad) with WMMA.
// Block = 128 threads = 4 waves. Macro tile: 64 output pixels x 64 Cout.
// Each wave owns one 16-Cout tile and 4 independent 16x16 accumulators
// (4 M sub-tiles), so one B-fragment load feeds 4 back-to-back WMMAs.
// A tile (64 px x 32 ch, 4 KB) is staged in LDS once per K-step.
//   act   : fp16 NHWC activations [N, IH, IW, Cin]
//   bfrag : fp16 ternary weights pre-packed in WMMA B-fragment layout:
//           [ (tap*CinTiles+ct)*CoutTiles + nt ][ lane(32) ][ half(16) ]
//   y     : f32 output [P, Cout] with bias fused, P = N*OH*OW
// ---------------------------------------------------------------------------
__global__ __launch_bounds__(128) void conv_wmma_k(
    const _Float16* __restrict__ act, const _Float16* __restrict__ bfrag,
    const float* __restrict__ bias, float* __restrict__ y,
    int N, int IH, int IW, int Cin, int OH, int OW, int Cout,
    int KH, int KW, int stride, int pad)
{
  __shared__ __align__(16) _Float16 As[64 * 32];   // 64 pixels x 32 channels

  const int lane = threadIdx.x & 31;
  const int wave = threadIdx.x >> 5;
  const int CoutTiles = (Cout + 15) >> 4;
  const int CinTiles  = (Cin + 31) >> 5;
  const int nt = blockIdx.y * 4 + wave;
  const bool wvalid = (nt < CoutTiles);
  const int mt = blockIdx.x;                       // 64-pixel macro tile
  const long P = (long)N * OH * OW;

  const int M = lane & 15;                         // A row within 16-row tile
  const int kbase = (lane >> 4) * 8;               // K sub-offset per lane group

  // cooperative fill: thread t loads 16 halves of pixel (t>>1)
  const int fpix = threadIdx.x >> 1;               // 0..63
  const int fc0  = (threadIdx.x & 1) * 16;         // 0 or 16
  const long fp  = (long)mt * 64 + fpix;
  const bool fvalid = fp < P;
  int fn = 0, foh = 0, fow = 0;
  if (fvalid) { fn = (int)(fp / (OH * OW)); int r = (int)(fp % (OH * OW)); foh = r / OW; fow = r % OW; }

  v8f acc[4];
  #pragma unroll
  for (int mi = 0; mi < 4; ++mi)
    #pragma unroll
    for (int r = 0; r < 8; ++r) acc[mi][r] = 0.0f;

  const int totalK = KH * KW * CinTiles;
  for (int kh = 0; kh < KH; ++kh) {
    const int fih = foh * stride - pad + kh;
    for (int kw = 0; kw < KW; ++kw) {
      const int fiw = fow * stride - pad + kw;
      const bool fok = fvalid && fih >= 0 && fih < IH && fiw >= 0 && fiw < IW;
      const _Float16* frow = act + ((long)(fn * IH + fih) * IW + fiw) * Cin;
      for (int ct = 0; ct < CinTiles; ++ct) {
        // ---- stage 64x32 A tile in LDS ----
        if (fok && (Cin & 31) == 0) {
          const v4u* src = (const v4u*)(frow + ct * 32 + fc0);
          v4u* dv = (v4u*)&As[fpix * 32 + fc0];
          dv[0] = src[0];
          dv[1] = src[1];
        } else {
          #pragma unroll
          for (int u = 0; u < 16; ++u) {
            const int cin = ct * 32 + fc0 + u;
            As[fpix * 32 + fc0 + u] = (fok && cin < Cin) ? frow[cin] : (_Float16)0.0f;
          }
        }
        __syncthreads();
        if (wvalid) {
          const int kstep = (kh * KW + kw) * CinTiles + ct;
          AFrag b;
          const v4u* bs = (const v4u*)(bfrag + ((long)kstep * CoutTiles + nt) * 512 + lane * 16);
          b.q[0] = bs[0];
          b.q[1] = bs[1];
          if (kstep + 1 < totalK)                          // L2-resident weights ->
            __builtin_prefetch(bfrag + ((long)(kstep + 1) * CoutTiles + nt) * 512 + lane * 16, 0, 1);
          #pragma unroll
          for (int mi = 0; mi < 4; ++mi) {                 // 4 independent WMMAs
            AFrag a;
            const v4u* ls = (const v4u*)&As[(mi * 16 + M) * 32 + kbase];
            a.q[0] = ls[0];                                // K = kbase + 0..7
            a.q[1] = ls[2];                                // K = kbase + 16..23
            acc[mi] = __builtin_amdgcn_wmma_f32_16x16x32_f16(
                          false, a.v, false, b.v, (short)0, acc[mi], false, false);
          }
        }
        __syncthreads();
      }
    }
  }

  if (wvalid) {
    const int co = nt * 16 + (lane & 15);
    if (co < Cout) {
      const float bb = bias ? bias[co] : 0.0f;
      #pragma unroll
      for (int mi = 0; mi < 4; ++mi)
        #pragma unroll
        for (int r = 0; r < 8; ++r) {                      // lanes16-31: M=r+8
          const long pr = (long)mt * 64 + mi * 16 + (lane >> 4) * 8 + r;
          if (pr < P) y[pr * Cout + co] = acc[mi][r] + bb;
        }
    }
  }
}

// ---------------------------------------------------------------------------
// 1x1 convolution (pure GEMM over [P, Cin]) with Tensor-Data-Mover A staging:
// the A tile is an exact 2D tile (64 rows x 32 halves, row stride Cin), so
// wave 0 issues one TENSOR_LOAD_TO_LDS per K-step (D# built per ISA 08 §8.3/8.4)
// and synchronizes with s_wait_tensorcnt + workgroup barrier.
// ---------------------------------------------------------------------------
__global__ __launch_bounds__(128) void conv1x1_wmma_tdm_k(
    const _Float16* __restrict__ act, const _Float16* __restrict__ bfrag,
    const float* __restrict__ bias, float* __restrict__ y,
    long P, int Cin, int Cout)
{
  __shared__ __align__(16) _Float16 As[64 * 32];

  const int lane = threadIdx.x & 31;
  const int wave = threadIdx.x >> 5;
  const int CoutTiles = (Cout + 15) >> 4;
  const int CinTiles  = (Cin + 31) >> 5;
  const int nt = blockIdx.y * 4 + wave;
  const bool wvalid = (nt < CoutTiles);
  const int mt = blockIdx.x;
  const long p0 = (long)mt * 64;

  const int M = lane & 15;
  const int kbase = (lane >> 4) * 8;
  const unsigned ldsOff = (unsigned)(size_t)(&As[0]);      // LDS byte address

  v8f acc[4];
  #pragma unroll
  for (int mi = 0; mi < 4; ++mi)
    #pragma unroll
    for (int r = 0; r < 8; ++r) acc[mi][r] = 0.0f;

  const long rowsLeft = P - p0;                            // >= 1 by grid sizing

  for (int ct = 0; ct < CinTiles; ++ct) {
    __syncthreads();                                       // As consumable again
    if (wave == 0) {
      // ---- build D# (groups 0..3) for a 2D tile: 32 ch x 64 rows ----
      const unsigned long long ga =
          (unsigned long long)(size_t)act +
          ((unsigned long long)(p0 * (long)Cin + (long)ct * 32)) * 2ull;
      const unsigned dim0 = (unsigned)(Cin - ct * 32);     // remaining channels
      const unsigned dim1 = (unsigned)hmin_l(rowsLeft, 0x7fffffffL);
      const unsigned long long s0 = (unsigned long long)(unsigned)Cin; // row stride

      tdm4u g0;
      g0[0] = 1u;                                          // count=1 (valid D#)
      g0[1] = ldsOff;                                      // lds_addr
      g0[2] = (unsigned)(ga & 0xFFFFFFFFu);                // global_addr[31:0]
      g0[3] = (unsigned)((ga >> 32) & 0x01FFFFFFu) | (2u << 30); // addr[56:32]|type=2

      tdm8i g1;
      g1[0] = (int)(1u << 16);                             // data_size=1 (2 bytes)
      g1[1] = (int)((dim0 & 0xFFFFu) << 16);               // tensor_dim0[15:0] @bit48
      g1[2] = (int)(((dim0 >> 16) & 0xFFFFu) | ((dim1 & 0xFFFFu) << 16));
      g1[3] = (int)(((dim1 >> 16) & 0xFFFFu) | (32u << 16)); // tile_dim0=32
      g1[4] = 64;                                          // tile_dim1=64, tile_dim2=0
      g1[5] = (int)(s0 & 0xFFFFFFFFull);                   // tensor_dim0_stride[31:0]
      g1[6] = (int)((s0 >> 32) & 0xFFFFull);               // stride[47:32]
      g1[7] = 0;                                           // tensor_dim1_stride=0 (2D)

      tdm4i g2 = {0, 0, 0, 0};
      tdm4i g3 = {0, 0, 0, 0};
#if defined(__clang_major__) && (__clang_major__ >= 23)
      tdm8i g4 = {0, 0, 0, 0, 0, 0, 0, 0};
      __builtin_amdgcn_tensor_load_to_lds(g0, g1, g2, g3, g4, 0);
#else
      __builtin_amdgcn_tensor_load_to_lds(g0, g1, g2, g3, 0);
#endif
      __builtin_amdgcn_s_wait_tensorcnt(0);                // DMA complete
    }
    __syncthreads();                                       // publish LDS tile

    if (wvalid) {
      AFrag b;
      const v4u* bs = (const v4u*)(bfrag + ((long)ct * CoutTiles + nt) * 512 + lane * 16);
      b.q[0] = bs[0];
      b.q[1] = bs[1];
      if (ct + 1 < CinTiles)
        __builtin_prefetch(bfrag + ((long)(ct + 1) * CoutTiles + nt) * 512 + lane * 16, 0, 1);
      #pragma unroll
      for (int mi = 0; mi < 4; ++mi) {
        AFrag a;
        const v4u* ls = (const v4u*)&As[(mi * 16 + M) * 32 + kbase];
        a.q[0] = ls[0];
        a.q[1] = ls[2];
        acc[mi] = __builtin_amdgcn_wmma_f32_16x16x32_f16(
                      false, a.v, false, b.v, (short)0, acc[mi], false, false);
      }
    }
  }
  __syncthreads();

  if (wvalid) {
    const int co = nt * 16 + (lane & 15);
    if (co < Cout) {
      const float bb = bias ? bias[co] : 0.0f;
      #pragma unroll
      for (int mi = 0; mi < 4; ++mi)
        #pragma unroll
        for (int r = 0; r < 8; ++r) {
          const long pr = p0 + mi * 16 + (lane >> 4) * 8 + r;
          if (pr < P) y[pr * Cout + co] = acc[mi][r] + bb;
        }
    }
  }
}

// ---------------------------------------------------------------------------
// BatchNorm batch statistics: one block per channel over P pixels ([P,C] f32).
// ---------------------------------------------------------------------------
__global__ __launch_bounds__(256) void bn_stats_k(
    const float* __restrict__ y, long P, int C,
    float* __restrict__ mean, float* __restrict__ invstd)
{
  __shared__ float s1[256], s2[256];
  const int ch = blockIdx.x;
  float a = 0.f, b = 0.f;
  for (long i = threadIdx.x; i < P; i += blockDim.x) {
    const float v = y[i * C + ch];
    a += v; b += v * v;
  }
  s1[threadIdx.x] = a; s2[threadIdx.x] = b;
  __syncthreads();
  for (int s = 128; s > 0; s >>= 1) {
    if ((int)threadIdx.x < s) { s1[threadIdx.x] += s1[threadIdx.x + s]; s2[threadIdx.x] += s2[threadIdx.x + s]; }
    __syncthreads();
  }
  if (threadIdx.x == 0) {
    const float m = s1[0] / (float)P;
    const float var = s2[0] / (float)P - m * m;
    mean[ch] = m;
    invstd[ch] = rsqrtf(fmaxf(var, 0.f) + BN_EPS);
  }
}

// ---------------------------------------------------------------------------
// Fused BN-apply (+ optional residual) (+ optional SiLU):
//  mode 0: silu(bn(y)) -> fp16 | mode 1: bn(y) -> f32
//  mode 2: silu(bn(y)+resF32) -> fp16 | mode 3: silu(bn(y)+resF16) -> fp16
// ---------------------------------------------------------------------------
__global__ __launch_bounds__(256) void bn_act_k(
    const float* __restrict__ y, const float* __restrict__ mean,
    const float* __restrict__ invstd, const float* __restrict__ g,
    const float* __restrict__ beta,
    const float* resF, const _Float16* resH,
    int mode, long total, int C, _Float16* outH, float* outF)
{
  for (long i = (long)blockIdx.x * blockDim.x + threadIdx.x; i < total;
       i += (long)gridDim.x * blockDim.x) {
    const int ch = (int)(i % C);
    float v = (y[i] - mean[ch]) * invstd[ch] * g[ch] + beta[ch];
    if (mode == 1) { outF[i] = v; continue; }
    if (mode == 2) v += resF[i];
    else if (mode == 3) v += (float)resH[i];
    v = v / (1.f + __expf(-v));
    outH[i] = (_Float16)v;
  }
}

// f32 NCHW input -> fp16 NHWC
__global__ __launch_bounds__(256) void nchw_to_nhwc_k(
    const float* __restrict__ x, _Float16* __restrict__ out, int N, int C, int H, int W)
{
  const long total = (long)N * C * H * W;
  for (long i = (long)blockIdx.x * blockDim.x + threadIdx.x; i < total;
       i += (long)gridDim.x * blockDim.x) {
    const int c = (int)(i % C); long t = i / C;
    const int w = (int)(t % W); t /= W;
    const int h = (int)(t % H);
    const int n = (int)(t / H);
    out[i] = (_Float16)x[((long)(n * C + c) * H + h) * W + w];
  }
}

// global average pool over HW, fp16 NHWC -> fp16 [N, C]
__global__ __launch_bounds__(256) void avgpool_k(
    const _Float16* __restrict__ act, int HW, int C, long NC, _Float16* __restrict__ out)
{
  for (long i = (long)blockIdx.x * blockDim.x + threadIdx.x; i < NC;
       i += (long)gridDim.x * blockDim.x) {
    const int c = (int)(i % C); const long n = i / C;
    float s = 0.f;
    for (int t = 0; t < HW; ++t) s += (float)act[(n * HW + t) * C + c];
    out[i] = (_Float16)(s / (float)HW);
  }
}

// ----- ternary quantization: median(|W|) via absmax + histogram select -----
__global__ __launch_bounds__(256) void absmax_k(
    const float* __restrict__ w, long n, unsigned* __restrict__ amax)
{
  __shared__ unsigned sm[256];
  unsigned m = 0u;
  for (long i = (long)blockIdx.x * blockDim.x + threadIdx.x; i < n;
       i += (long)gridDim.x * blockDim.x) {
    const unsigned b = __float_as_uint(fabsf(w[i]));
    m = b > m ? b : m;
  }
  sm[threadIdx.x] = m; __syncthreads();
  for (int s = 128; s > 0; s >>= 1) {
    if ((int)threadIdx.x < s) { const unsigned o = sm[threadIdx.x + s]; if (o > sm[threadIdx.x]) sm[threadIdx.x] = o; }
    __syncthreads();
  }
  if (threadIdx.x == 0) atomicMax(amax, sm[0]);
}

__global__ __launch_bounds__(256) void hist_k(
    const float* __restrict__ w, long n, const unsigned* __restrict__ amax,
    unsigned* __restrict__ hist)
{
  const float mx = __uint_as_float(*amax);
  const float inv = mx > 0.f ? (float)HBINS / mx : 0.f;
  for (long i = (long)blockIdx.x * blockDim.x + threadIdx.x; i < n;
       i += (long)gridDim.x * blockDim.x) {
    int b = (int)(fabsf(w[i]) * inv);
    if (b > HBINS - 1) b = HBINS - 1;
    atomicAdd(&hist[b], 1u);
  }
}

__global__ void median_k(const unsigned* __restrict__ hist,
                         const unsigned* __restrict__ amax, long n,
                         float* __restrict__ scale)
{
  if (threadIdx.x == 0 && blockIdx.x == 0) {
    const float mx = __uint_as_float(*amax);
    const long half = (n + 1) / 2;
    long cum = 0; int b = 0;
    for (; b < HBINS; ++b) { cum += hist[b]; if (cum >= half) break; }
    const float s = ((float)b + 0.5f) * mx / (float)HBINS;
    *scale = fmaxf(s, 1e-8f);
  }
}

// pack ternary fp16 weights straight into the WMMA B-fragment layout
__global__ __launch_bounds__(256) void pack_k(
    const float* __restrict__ w, const float* __restrict__ scale,
    int Cin, int Cout, int KH, int KW, _Float16* __restrict__ dst)
{
  const int CinTiles = (Cin + 31) / 32, CoutTiles = (Cout + 15) / 16;
  const long total = (long)KH * KW * CinTiles * CoutTiles * 512;
  const float s = *scale;
  for (long i = (long)blockIdx.x * blockDim.x + threadIdx.x; i < total;
       i += (long)gridDim.x * blockDim.x) {
    const int e = (int)(i & 15);
    const int lane = (int)((i >> 4) & 31);
    const long rest = i >> 9;
    const int nt = (int)(rest % CoutTiles);
    const long ks = rest / CoutTiles;
    const int ct = (int)(ks % CinTiles);
    const int tap = (int)(ks / CinTiles);
    const int kh = tap / KW, kw = tap % KW;
    const int K = ((lane >> 4) & 1) * 8 + (e & 7) + (e >> 3) * 16;
    const int cin = ct * 32 + K;
    const int co  = nt * 16 + (lane & 15);
    float q = 0.f;
    if (cin < Cin && co < Cout) {
      const float v = w[((long)(co * Cin + cin) * KH + kh) * KW + kw];  // OIHW
      q = fminf(1.f, fmaxf(-1.f, rintf(v / s))) * s;
    }
    dst[i] = (_Float16)q;
  }
}

// ---------------------------------------------------------------------------
// Host-side orchestration
// ---------------------------------------------------------------------------
extern "C" void kernel_launch(void* const* d_in, const int* in_sizes, int n_in,
                              void* d_out, int out_size, void* d_ws, size_t ws_size,
                              hipStream_t stream)
{
  (void)in_sizes; (void)n_in; (void)ws_size;
  static const int PLANES[4] = {64, 128, 256, 512};
  static const int NBLK[4]   = {3, 4, 6, 3};
  static const int STRIDE[4] = {1, 2, 2, 2};
  const int BATCH = 128;

  int cur = 0;
  const float* X         = (const float*)d_in[cur++];
  const float* stem_w    = (const float*)d_in[cur++];
  const float* stem_bias = (const float*)d_in[cur++];
  const float* stem_g    = (const float*)d_in[cur++];
  const float* stem_beta = (const float*)d_in[cur++];

  struct QW { const float* w; int Cout, Cin, KH, KW; size_t off; };
  QW qws[64]; int nq = 0; size_t qoff = 0;
  auto addq = [&](const float* w, int Cout, int Cin, int KH, int KW) -> int {
    QW q; q.w = w; q.Cout = Cout; q.Cin = Cin; q.KH = KH; q.KW = KW; q.off = qoff;
    qoff += (size_t)KH * KW * ((Cin + 31) / 32) * ((Cout + 15) / 16) * 512;
    qws[nq] = q; return nq++;
  };
  const int qstem = addq(stem_w, 64, 3, 3, 3);

  struct Blk {
    const float *c1b, *g1, *b1, *c2b, *g2, *b2, *c3b, *g3, *b3;
    const float *dsb, *dsg, *dsbb;
    bool hasDs; int planes, stride, inpl, outc;
    int q1, q2, q3, qd;
  } blks[16];
  int nb = 0, inpl = 64;
  for (int st = 0; st < 4; ++st) {
    for (int j = 0; j < NBLK[st]; ++j) {
      Blk& B = blks[nb++];
      B.planes = PLANES[st]; B.stride = (j == 0) ? STRIDE[st] : 1;
      B.inpl = inpl; B.outc = PLANES[st] * 4;
      const float* c1w = (const float*)d_in[cur++]; B.c1b = (const float*)d_in[cur++];
      B.g1 = (const float*)d_in[cur++]; B.b1 = (const float*)d_in[cur++];
      const float* c2w = (const float*)d_in[cur++]; B.c2b = (const float*)d_in[cur++];
      B.g2 = (const float*)d_in[cur++]; B.b2 = (const float*)d_in[cur++];
      const float* c3w = (const float*)d_in[cur++]; B.c3b = (const float*)d_in[cur++];
      B.g3 = (const float*)d_in[cur++]; B.b3 = (const float*)d_in[cur++];
      B.hasDs = (j == 0) && (B.stride != 1 || inpl != B.outc);
      const float* dsw = nullptr;
      if (B.hasDs) {
        dsw = (const float*)d_in[cur++]; B.dsb = (const float*)d_in[cur++];
        B.dsg = (const float*)d_in[cur++]; B.dsbb = (const float*)d_in[cur++];
      } else { B.dsb = B.dsg = B.dsbb = nullptr; }
      B.q1 = addq(c1w, B.planes, inpl, 1, 1);
      B.q2 = addq(c2w, B.planes, B.planes, 3, 3);
      B.q3 = addq(c3w, B.outc, B.planes, 1, 1);
      B.qd = B.hasDs ? addq(dsw, B.outc, inpl, 1, 1) : -1;
      inpl = B.outc;
    }
  }
  int NCLS = out_size / BATCH; if (NCLS <= 0) NCLS = 100;
  const float* head_w = (const float*)d_in[cur++];
  const float* head_b = (const float*)d_in[cur++];
  const int qhead = addq(head_w, NCLS, 2048, 1, 1);

  // ---- workspace bump allocator ----
  char* wp = (char*)d_ws;
  auto walloc = [&](size_t bytes) -> void* {
    void* r = (void*)wp; wp += (bytes + 255) & ~(size_t)255; return r;
  };
  float*    scales = (float*)   walloc(64 * sizeof(float));
  unsigned* amax   = (unsigned*)walloc(64 * sizeof(unsigned));
  unsigned* hist   = (unsigned*)walloc((size_t)64 * HBINS * sizeof(unsigned));
  const size_t statBytes = (size_t)(wp - (char*)scales);
  float*    bnMean = (float*)walloc(2048 * sizeof(float));
  float*    bnInv  = (float*)walloc(2048 * sizeof(float));
  _Float16* qw     = (_Float16*)walloc(qoff * sizeof(_Float16));
  const size_t MAXACT = (size_t)BATCH * 32 * 32 * 256;
  _Float16* T0 = (_Float16*)walloc(MAXACT * 2);
  _Float16* T1 = (_Float16*)walloc(MAXACT * 2);
  _Float16* T2 = (_Float16*)walloc(MAXACT * 2);
  float*    Y  = (float*)walloc(MAXACT * 4);
  float*    D  = (float*)walloc(MAXACT * 4);

  (void)hipMemsetAsync(scales, 0, statBytes, stream);

  // ---- quantize + pack every ternary weight tensor ----
  for (int i = 0; i < nq; ++i) {
    const QW& q = qws[i];
    const long n = (long)q.Cout * q.Cin * q.KH * q.KW;
    const int rb = (int)hmin_l((n + 255) / 256, 2048);
    absmax_k<<<rb, 256, 0, stream>>>(q.w, n, &amax[i]);
    hist_k<<<rb, 256, 0, stream>>>(q.w, n, &amax[i], &hist[(size_t)i * HBINS]);
    median_k<<<1, 32, 0, stream>>>(&hist[(size_t)i * HBINS], &amax[i], n, &scales[i]);
    const long tot = (long)q.KH * q.KW * ((q.Cin + 31) / 32) * ((q.Cout + 15) / 16) * 512;
    const int pb = (int)hmin_l((tot + 255) / 256, 4096);
    pack_k<<<pb, 256, 0, stream>>>(q.w, &scales[i], q.Cin, q.Cout, q.KH, q.KW, qw + q.off);
  }

  // ---- helpers ----
  auto conv = [&](const _Float16* a, int qi, const float* bias, float* yo,
                  int IH, int IW, int Cin, int Cout, int K, int stride, int pad,
                  int& OH, int& OW) {
    OH = (IH + 2 * pad - K) / stride + 1;
    OW = (IW + 2 * pad - K) / stride + 1;
    const long P = (long)BATCH * OH * OW;
    const int CoutTiles = (Cout + 15) / 16;
    dim3 grid((unsigned)((P + 63) / 64), (unsigned)((CoutTiles + 3) / 4));
    if (K == 1 && stride == 1 && pad == 0) {
      conv1x1_wmma_tdm_k<<<grid, 128, 0, stream>>>(a, qw + qws[qi].off, bias, yo,
                                                   P, Cin, Cout);
    } else {
      conv_wmma_k<<<grid, 128, 0, stream>>>(a, qw + qws[qi].off, bias, yo,
                                            BATCH, IH, IW, Cin, OH, OW, Cout,
                                            K, K, stride, pad);
    }
  };
  auto bn = [&](float* yv, long P, int C, const float* g_, const float* be,
                int mode, const float* resF, const _Float16* resH,
                _Float16* oH, float* oF) {
    bn_stats_k<<<C, 256, 0, stream>>>(yv, P, C, bnMean, bnInv);
    const long total = P * (long)C;
    const int blocks = (int)hmin_l((total + 255) / 256, 8192);
    bn_act_k<<<blocks, 256, 0, stream>>>(yv, bnMean, bnInv, g_, be, resF, resH,
                                         mode, total, C, oH, oF);
  };

  // ---- forward pass ----
  nchw_to_nhwc_k<<<1024, 256, 0, stream>>>(X, T0, BATCH, 3, 32, 32);

  int H = 32, W = 32, C = 64, OH, OW;
  conv(T0, qstem, stem_bias, Y, 32, 32, 3, 64, 3, 1, 1, OH, OW);
  bn(Y, (long)BATCH * OH * OW, 64, stem_g, stem_beta, 0, nullptr, nullptr, T0, nullptr);
  H = OH; W = OW; C = 64;

  for (int b = 0; b < nb; ++b) {
    Blk& B = blks[b];
    int OH1, OW1, OH2, OW2, OH3, OW3;
    conv(T0, B.q1, B.c1b, Y, H, W, C, B.planes, 1, 1, 0, OH1, OW1);
    bn(Y, (long)BATCH * OH1 * OW1, B.planes, B.g1, B.b1, 0, nullptr, nullptr, T1, nullptr);
    conv(T1, B.q2, B.c2b, Y, OH1, OW1, B.planes, B.planes, 3, B.stride, 1, OH2, OW2);
    bn(Y, (long)BATCH * OH2 * OW2, B.planes, B.g2, B.b2, 0, nullptr, nullptr, T2, nullptr);
    conv(T2, B.q3, B.c3b, Y, OH2, OW2, B.planes, B.outc, 1, 1, 0, OH3, OW3);
    const long P3 = (long)BATCH * OH3 * OW3;
    if (B.hasDs) {
      int dH, dW;
      conv(T0, B.qd, B.dsb, D, H, W, C, B.outc, 1, B.stride, 0, dH, dW);
      bn(D, P3, B.outc, B.dsg, B.dsbb, 1, nullptr, nullptr, nullptr, D);
      bn(Y, P3, B.outc, B.g3, B.b3, 2, D, nullptr, T0, nullptr);
    } else {
      bn(Y, P3, B.outc, B.g3, B.b3, 3, nullptr, T0, T0, nullptr);
    }
    H = OH3; W = OW3; C = B.outc;
  }

  avgpool_k<<<1024, 256, 0, stream>>>(T0, H * W, C, (long)BATCH * C, T1);
  int oh, ow;
  conv(T1, qhead, head_b, (float*)d_out, 1, 1, C, NCLS, 1, 1, 0, oh, ow);
}
